// PointCloudToGaussian_66984309948587
// MI455X (gfx1250) — matrix operand order repaired and checked by reading, not compiled, not run on hardware
//
#include <hip/hip_runtime.h>
#include <math.h>

// ---------------- problem constants (from reference) ----------------
#define BB   32          // batches
#define NIN  8192        // input points per batch
#define MS   819         // FPS samples = NIN * 0.1
#define NP   1638        // 2*MS points after jitter-concat
#define KK   16          // K neighbors
#define HID  64          // MLP hidden
#define EPSF 1e-6f

#define KT   7           // ceil(NP/256) KNN tiles
#define MT   103         // ceil(NP/16) WMMA row-tiles

// output layout: (pts, cov, opacity, R, scale) concatenated flat
#define PTS_OFF 0
#define COV_OFF (BB*NP*3)
#define OPA_OFF (COV_OFF + BB*NP*9)
#define R_OFF   (OPA_OFF + BB*NP)
#define SCL_OFF (R_OFF   + BB*NP*9)

typedef __attribute__((ext_vector_type(16))) _Float16 v16h;
typedef __attribute__((ext_vector_type(8)))  float    v8f;

// gfx1250 async global->LDS path (ASYNCcnt-tracked), guarded so absence of the
// builtin on a given toolchain falls back to plain loads without breaking.
#if __has_builtin(__builtin_amdgcn_global_load_async_to_lds_b32)
#define HAVE_ASYNC_LDS 1
#else
#define HAVE_ASYNC_LDS 0
#endif

__device__ inline void wait_asynccnt0(){
#if __has_builtin(__builtin_amdgcn_s_wait_asynccnt)
  __builtin_amdgcn_s_wait_asynccnt(0);
#else
  asm volatile("s_wait_asynccnt 0" ::: "memory");
#endif
}

// ---------------- small helpers ----------------
__device__ inline unsigned hashu(unsigned x){
  x ^= x >> 16; x *= 0x7feb352dU; x ^= x >> 15; x *= 0x846ca68bU; x ^= x >> 16;
  return x;
}
// hash-based standard normal (Box-Muller); JAX PRNG not bit-reproducible
__device__ inline float gnoise(unsigned s){
  unsigned a = hashu(s * 0x9E3779B9u + 1u);
  unsigned c = hashu(s ^ 0x85ebca6bu);
  float u1 = ((a >> 8) + 1u) * (1.0f/16777216.0f);
  float u2 = (c >> 8) * (1.0f/16777216.0f);
  return sqrtf(-2.f*logf(u1)) * cosf(6.28318530718f*u2);
}
__device__ inline float gelu_exact(float x){
  return 0.5f * x * (1.f + erff(x * 0.70710678118f));
}

// A-fragment element -> K index (contiguous order k = 16*hi + e, matching B fills)
__device__ inline int a_kidx(int lane, int e){
  int hi = (lane >> 4) & 1;
  return hi*16 + e;
}

// ================= kernel 1: farthest point sampling + jittered pts =========
__global__ __launch_bounds__(1024)
void fps_kernel(const float* __restrict__ points, int* __restrict__ idx,
                float* __restrict__ pts)
{
  const int b = blockIdx.x;
  const int t = threadIdx.x;
  __shared__ float s_last[3];
  __shared__ float rv[32];
  __shared__ int   ri[32];
  __shared__ int   s_cur;

  const float* pb = points + (size_t)b * NIN * 3;
  __builtin_prefetch(pb + t * 3, 0, 1);   // -> global_prefetch_b8

  // register-resident points (strided ownership: point p owned by thread p&1023)
  float px[8], py[8], pz[8], mind[8];
#pragma unroll
  for (int k = 0; k < 8; k++){
    int p = t + k * 1024;
    px[k] = pb[p*3+0]; py[k] = pb[p*3+1]; pz[k] = pb[p*3+2];
    mind[k] = 1e10f;
  }
  if (t == 0){ s_cur = 0; idx[b*MS + 0] = 0; }
  __syncthreads();

  for (int i = 1; i < MS; i++){
    int cur = s_cur;
    if (t == 0){
      s_last[0] = pb[cur*3+0]; s_last[1] = pb[cur*3+1]; s_last[2] = pb[cur*3+2];
    }
    __syncthreads();
    float lx = s_last[0], ly = s_last[1], lz = s_last[2];
    float bv = -1.f; int bi = 0;
#pragma unroll
    for (int k = 0; k < 8; k++){
      float dx = px[k]-lx, dy = py[k]-ly, dz = pz[k]-lz;
      float d = dx*dx + dy*dy + dz*dz;
      mind[k] = fminf(mind[k], d);
      int p = t + k * 1024;                // ascending -> first-max tiebreak
      if (mind[k] > bv){ bv = mind[k]; bi = p; }
    }
    // wave32 argmax (first-index tiebreak)
    for (int m2 = 16; m2 > 0; m2 >>= 1){
      float ov = __shfl_xor(bv, m2, 32);
      int   oi = __shfl_xor(bi, m2, 32);
      if (ov > bv || (ov == bv && oi < bi)){ bv = ov; bi = oi; }
    }
    if ((t & 31) == 0){ rv[t >> 5] = bv; ri[t >> 5] = bi; }
    __syncthreads();
    if (t < 32){
      bv = rv[t]; bi = ri[t];
      for (int m2 = 16; m2 > 0; m2 >>= 1){
        float ov = __shfl_xor(bv, m2, 32);
        int   oi = __shfl_xor(bi, m2, 32);
        if (ov > bv || (ov == bv && oi < bi)){ bv = ov; bi = oi; }
      }
      if (t == 0){ s_cur = bi; idx[b*MS + i] = bi; }
    }
    __syncthreads();
  }

  // pts = concat(sampled, sampled + 0.01*noise)
  for (int j = t; j < MS; j += 1024){
    int id = idx[b*MS + j];
    float x = pb[id*3+0], y = pb[id*3+1], z = pb[id*3+2];
    float* o0 = pts + ((size_t)b*NP + j) * 3;
    float* o1 = pts + ((size_t)b*NP + MS + j) * 3;
    o0[0] = x; o0[1] = y; o0[2] = z;
    unsigned base = ((unsigned)(b*NP + MS + j)) * 3u ^ 0x2a2a2a2au;
    o1[0] = x + 0.01f * gnoise(base + 0u);
    o1[1] = y + 0.01f * gnoise(base + 1u);
    o1[2] = z + 0.01f * gnoise(base + 2u);
  }
}

// ================= kernel 2: brute-force KNN (top-17, drop self) ============
// Batch points staged into LDS with gfx1250 async global->LDS (ASYNCcnt) when
// available. AoS stage: query reads are stride-3 dwords (coprime with 64 banks
// -> conflict-free); candidate-loop reads are wave-uniform (broadcast).
__global__ __launch_bounds__(256)
void knn_kernel(const float* __restrict__ pts, int* __restrict__ knn)
{
  __shared__ float spts[NP*3];
  int b = blockIdx.x / KT, tile = blockIdx.x % KT;
  int t = threadIdx.x;
  const float* pb = pts + (size_t)b * NP * 3;

#if HAVE_ASYNC_LDS
  for (int j = t; j < NP*3; j += 256){
    __builtin_amdgcn_global_load_async_to_lds_b32(
        (int*)(pb + j), (int*)(spts + j), 0, 0);
  }
  wait_asynccnt0();
#else
  for (int j = t; j < NP*3; j += 256) spts[j] = pb[j];
#endif
  __syncthreads();

  int q = tile * 256 + t;
  if (q >= NP) return;
  float qx = spts[q*3+0], qy = spts[q*3+1], qz = spts[q*3+2];
  float bd[KK+1]; int bi[KK+1];
#pragma unroll
  for (int k = 0; k <= KK; k++){ bd[k] = 3.0e38f; bi[k] = -1; }
  for (int j = 0; j < NP; j++){
    float dx = spts[j*3+0]-qx, dy = spts[j*3+1]-qy, dz = spts[j*3+2]-qz;
    float d = dx*dx + dy*dy + dz*dz;
    if (d < bd[KK]){
      int pos = KK;
      while (pos > 0 && bd[pos-1] > d){ bd[pos] = bd[pos-1]; bi[pos] = bi[pos-1]; pos--; }
      bd[pos] = d; bi[pos] = j;
    }
  }
  int* o = knn + ((size_t)b*NP + q) * KK;
#pragma unroll
  for (int k = 0; k < KK; k++) o[k] = bi[k+1];   // drop nearest (self)
}

// ====== kernel 3: neighbor mean, 3x3 cov eigen -> linearness, density =======
__global__ __launch_bounds__(256)
void geom_kernel(const float* __restrict__ pts, const int* __restrict__ knn,
                 float* __restrict__ wc, float* __restrict__ lin,
                 float* __restrict__ den)
{
  int g = blockIdx.x * 256 + threadIdx.x;
  if (g >= BB*NP) return;
  int b = g / NP, q = g % NP;
  const float* pb = pts + (size_t)b * NP * 3;
  float cx = pb[q*3+0], cy = pb[q*3+1], cz = pb[q*3+2];
  const int* nb = knn + (size_t)g * KK;
  float nx[KK], ny[KK], nz[KK];
  float mx = 0.f, my = 0.f, mz = 0.f;
#pragma unroll
  for (int k = 0; k < KK; k++){
    int id = nb[k];
    nx[k] = pb[id*3+0]; ny[k] = pb[id*3+1]; nz[k] = pb[id*3+2];
    mx += nx[k]; my += ny[k]; mz += nz[k];
  }
  mx *= (1.f/KK); my *= (1.f/KK); mz *= (1.f/KK);
  wc[(size_t)g*3+0] = mx; wc[(size_t)g*3+1] = my; wc[(size_t)g*3+2] = mz;

  float a00=0,a01=0,a02=0,a11=0,a12=0,a22=0; int cnt = 0;
#pragma unroll
  for (int k = 0; k < KK; k++){
    float ex = nx[k]-cx, ey = ny[k]-cy, ez = nz[k]-cz;
    float d2 = ex*ex + ey*ey + ez*ez;
    cnt += (d2 < 0.01f) ? 1 : 0;           // DENSITY_RADIUS^2
    a00 += ex*ex; a01 += ex*ey; a02 += ex*ez;
    a11 += ey*ey; a12 += ey*ez; a22 += ez*ez;
  }
  float inv = 1.f / (15.f + EPSF);         // K-1+EPS
  a00*=inv; a01*=inv; a02*=inv; a11*=inv; a12*=inv; a22*=inv;

  // closed-form symmetric 3x3 eigenvalues (= singular values, PSD)
  float qd = (a00 + a11 + a22) * (1.f/3.f);
  float p1 = a01*a01 + a02*a02 + a12*a12;
  float p2 = (a00-qd)*(a00-qd) + (a11-qd)*(a11-qd) + (a22-qd)*(a22-qd) + 2.f*p1;
  float p  = sqrtf(fmaxf(p2, 0.f) * (1.f/6.f));
  float e0, e1, e2;
  if (p < 1e-12f){ e0 = e1 = e2 = qd; }
  else {
    float ip = 1.f / p;
    float b00=(a00-qd)*ip, b11=(a11-qd)*ip, b22=(a22-qd)*ip;
    float b01=a01*ip, b02=a02*ip, b12=a12*ip;
    float detB = b00*(b11*b22 - b12*b12) - b01*(b01*b22 - b12*b02)
               + b02*(b01*b12 - b11*b02);
    float r = fminf(fmaxf(detB * 0.5f, -1.f), 1.f);
    float phi = acosf(r) * (1.f/3.f);
    e0 = qd + 2.f*p*cosf(phi);
    e2 = qd + 2.f*p*cosf(phi + 2.09439510239f);
    e1 = 3.f*qd - e0 - e2;
  }
  lin[g] = e0 / (e1 + e2 + EPSF);
  den[g] = cnt * (1.f/KK);
}

// ====== kernel 4: fused rotation+scale MLPs on WMMA, then R/scale/cov =======
__global__ __launch_bounds__(32)
void mlp_wmma_kernel(const float* __restrict__ wc, const float* __restrict__ pts,
                     const float* __restrict__ rw1, const float* __restrict__ rb1,
                     const float* __restrict__ rw2, const float* __restrict__ rb2,
                     const float* __restrict__ sw1, const float* __restrict__ sb1,
                     const float* __restrict__ sw2, const float* __restrict__ sb2,
                     float* __restrict__ outCov, float* __restrict__ outR,
                     float* __restrict__ outS)
{
  int b = blockIdx.x / MT, tile = blockIdx.x % MT;
  int lane = threadIdx.x;
  int n16 = lane & 15, hi = lane >> 4;
  int row0 = tile * 16;

  __shared__ float hbuf[16][HID];   // hidden activations (transpose bounce)
  __shared__ float aab[16][16];     // layer-2 output tile
  __shared__ float Rb[16][9];       // per-row rotation matrix

  // ---------------- rotation path: A = weighted_center rows ----------------
  v16h aIn = {};
#pragma unroll
  for (int e = 0; e < 16; e++){
    int Kk = a_kidx(lane, e);
    int grow = row0 + n16;
    float v = (Kk < 3 && grow < NP) ? wc[((size_t)b*NP + grow)*3 + Kk] : 0.f;
    aIn[e] = (_Float16)v;
  }
#pragma unroll
  for (int tcol = 0; tcol < 4; tcol++){
    v16h bw = {};
#pragma unroll
    for (int e = 0; e < 16; e++){
      int Kk = hi*16 + e;
      int nn = tcol*16 + n16;
      bw[e] = (_Float16)((Kk < 3) ? rw1[Kk*HID + nn] : 0.f);
    }
    v8f c = {};
    c = __builtin_amdgcn_wmma_f32_16x16x32_f16(false, aIn, false, bw,
                                               (short)0, c, false, false);
    float bias = rb1[tcol*16 + n16];
#pragma unroll
    for (int r = 0; r < 8; r++)
      hbuf[r + hi*8][tcol*16 + n16] = gelu_exact(c[r] + bias);
  }
  __syncthreads();
  {
    v16h a0 = {}, a1 = {}, b0 = {}, b1 = {};
#pragma unroll
    for (int e = 0; e < 16; e++){
      int Kk = a_kidx(lane, e);
      a0[e] = (_Float16)hbuf[n16][Kk];
      a1[e] = (_Float16)hbuf[n16][Kk + 32];
      int Kb = hi*16 + e;
      b0[e] = (_Float16)((n16 < 3) ? rw2[Kb*3 + n16] : 0.f);
      b1[e] = (_Float16)((n16 < 3) ? rw2[(Kb+32)*3 + n16] : 0.f);
    }
    v8f c2 = {};
    c2 = __builtin_amdgcn_wmma_f32_16x16x32_f16(false, a0, false, b0,
                                                (short)0, c2, false, false);
    c2 = __builtin_amdgcn_wmma_f32_16x16x32_f16(false, a1, false, b1,
                                                (short)0, c2, false, false);
#pragma unroll
    for (int r = 0; r < 8; r++)
      aab[r + hi*8][n16] = c2[r] + ((n16 < 3) ? rb2[n16] : 0.f);
  }
  __syncthreads();
  if (lane < 16){
    float ax = aab[lane][0], ay = aab[lane][1], az = aab[lane][2];
    float ang = sqrtf(ax*ax + ay*ay + az*az);
    float R[9];
    if (ang < 1e-4f){
      R[0]=1;R[1]=0;R[2]=0; R[3]=0;R[4]=1;R[5]=0; R[6]=0;R[7]=0;R[8]=1;
    } else {
      float sa = fminf(ang, 3.14f);
      float iv = 1.f / (ang + EPSF);
      float x = ax*iv, y = ay*iv, z = az*iv;
      float ca = cosf(sa), sn = sinf(sa), C = 1.f - ca;
      R[0]=ca+x*x*C;    R[1]=x*y*C - z*sn; R[2]=x*z*C + y*sn;
      R[3]=y*x*C + z*sn;R[4]=ca+y*y*C;     R[5]=y*z*C - x*sn;
      R[6]=z*x*C - y*sn;R[7]=z*y*C + x*sn; R[8]=ca+z*z*C;
    }
#pragma unroll
    for (int k2 = 0; k2 < 9; k2++) Rb[lane][k2] = R[k2];
    int grow = row0 + lane;
    if (grow < NP){
      float* o = outR + ((size_t)b*NP + grow) * 9;
#pragma unroll
      for (int k2 = 0; k2 < 9; k2++) o[k2] = R[k2];
    }
  }
  __syncthreads();

  // ---------------- scale path: A = pts rows ----------------
  v16h aS = {};
#pragma unroll
  for (int e = 0; e < 16; e++){
    int Kk = a_kidx(lane, e);
    int grow = row0 + n16;
    float v = (Kk < 3 && grow < NP) ? pts[((size_t)b*NP + grow)*3 + Kk] : 0.f;
    aS[e] = (_Float16)v;
  }
#pragma unroll
  for (int tcol = 0; tcol < 4; tcol++){
    v16h bw = {};
#pragma unroll
    for (int e = 0; e < 16; e++){
      int Kk = hi*16 + e;
      int nn = tcol*16 + n16;
      bw[e] = (_Float16)((Kk < 3) ? sw1[Kk*HID + nn] : 0.f);
    }
    v8f c = {};
    c = __builtin_amdgcn_wmma_f32_16x16x32_f16(false, aS, false, bw,
                                               (short)0, c, false, false);
    float bias = sb1[tcol*16 + n16];
#pragma unroll
    for (int r = 0; r < 8; r++)
      hbuf[r + hi*8][tcol*16 + n16] = gelu_exact(c[r] + bias);
  }
  __syncthreads();
  {
    v16h a0 = {}, a1 = {}, b0 = {}, b1 = {};
#pragma unroll
    for (int e = 0; e < 16; e++){
      int Kk = a_kidx(lane, e);
      a0[e] = (_Float16)hbuf[n16][Kk];
      a1[e] = (_Float16)hbuf[n16][Kk + 32];
      int Kb = hi*16 + e;
      b0[e] = (_Float16)((n16 < 3) ? sw2[Kb*3 + n16] : 0.f);
      b1[e] = (_Float16)((n16 < 3) ? sw2[(Kb+32)*3 + n16] : 0.f);
    }
    v8f c2 = {};
    c2 = __builtin_amdgcn_wmma_f32_16x16x32_f16(false, a0, false, b0,
                                                (short)0, c2, false, false);
    c2 = __builtin_amdgcn_wmma_f32_16x16x32_f16(false, a1, false, b1,
                                                (short)0, c2, false, false);
#pragma unroll
    for (int r = 0; r < 8; r++)
      aab[r + hi*8][n16] = c2[r] + ((n16 < 3) ? sb2[n16] : 0.f);
  }
  __syncthreads();
  if (lane < 16){
    int grow = row0 + lane;
    if (grow < NP){
      float s[3];
#pragma unroll
      for (int c = 0; c < 3; c++){
        float x = aab[lane][c];
        float sp = (x > 20.f) ? x : log1pf(expf(x));       // softplus
        s[c] = fminf(fmaxf(sp, 0.01f), 0.5f);              // clip SC_LO..SC_HI
      }
      float* os = outS + ((size_t)b*NP + grow) * 3;
      os[0] = s[0]; os[1] = s[1]; os[2] = s[2];
      float s2[3] = { s[0]*s[0], s[1]*s[1], s[2]*s[2] };
      float* oc = outCov + ((size_t)b*NP + grow) * 9;
#pragma unroll
      for (int i = 0; i < 3; i++)
#pragma unroll
        for (int j = 0; j < 3; j++){
          float acc = 0.f;
#pragma unroll
          for (int k2 = 0; k2 < 3; k2++)
            acc += Rb[lane][i*3+k2] * s2[k2] * Rb[lane][j*3+k2];
          oc[i*3+j] = acc;
        }
    }
  }
}

// ================= kernel 5: per-batch min/max of linearness & density =====
__global__ __launch_bounds__(256)
void minmax_kernel(const float* __restrict__ lin, const float* __restrict__ den,
                   float* __restrict__ mm)
{
  int b = blockIdx.x, t = threadIdx.x;
  __shared__ float s0[256], s1[256], s2[256], s3[256];
  float lmin = 3e38f, lmax = -3e38f, dmin = 3e38f, dmax = -3e38f;
  for (int j = t; j < NP; j += 256){
    float l = lin[b*NP + j], d = den[b*NP + j];
    lmin = fminf(lmin, l); lmax = fmaxf(lmax, l);
    dmin = fminf(dmin, d); dmax = fmaxf(dmax, d);
  }
  s0[t]=lmin; s1[t]=lmax; s2[t]=dmin; s3[t]=dmax;
  __syncthreads();
  for (int m2 = 128; m2 > 0; m2 >>= 1){
    if (t < m2){
      s0[t]=fminf(s0[t],s0[t+m2]); s1[t]=fmaxf(s1[t],s1[t+m2]);
      s2[t]=fminf(s2[t],s2[t+m2]); s3[t]=fmaxf(s3[t],s3[t+m2]);
    }
    __syncthreads();
  }
  if (t == 0){
    mm[b*4+0]=s0[0]; mm[b*4+1]=s1[0]; mm[b*4+2]=s2[0]; mm[b*4+3]=s3[0];
  }
}

// ================= kernel 6: opacity ========================================
__global__ __launch_bounds__(256)
void opacity_kernel(const float* __restrict__ lin, const float* __restrict__ den,
                    const float* __restrict__ mm, float* __restrict__ opa)
{
  int g = blockIdx.x * 256 + threadIdx.x;
  if (g >= BB*NP) return;
  int b = g / NP;
  float ln = (lin[g] - mm[b*4+0]) / (mm[b*4+1] - mm[b*4+0] + EPSF);
  float dn = (den[g] - mm[b*4+2]) / (mm[b*4+3] - mm[b*4+2] + EPSF);
  dn = fmaxf(dn, 0.f);
  float c = ln * powf(dn, 0.45f);
  opa[g] = fminf(fmaxf(c, 0.05f), 0.95f);
}

// ================= host launch ==============================================
extern "C" void kernel_launch(void* const* d_in, const int* in_sizes, int n_in,
                              void* d_out, int out_size, void* d_ws, size_t ws_size,
                              hipStream_t stream)
{
  (void)in_sizes; (void)n_in; (void)out_size; (void)ws_size;
  const float* points = (const float*)d_in[0];
  const float* rw1 = (const float*)d_in[1];
  const float* rb1 = (const float*)d_in[2];
  const float* rw2 = (const float*)d_in[3];
  const float* rb2 = (const float*)d_in[4];
  const float* sw1 = (const float*)d_in[5];
  const float* sb1 = (const float*)d_in[6];
  const float* sw2 = (const float*)d_in[7];
  const float* sb2 = (const float*)d_in[8];

  float* out = (float*)d_out;
  float* pts = out + PTS_OFF;
  float* cov = out + COV_OFF;
  float* opa = out + OPA_OFF;
  float* Rm  = out + R_OFF;
  float* scl = out + SCL_OFF;

  char* ws = (char*)d_ws;
  int*   idx = (int*)ws;    ws += sizeof(int)   * (size_t)BB * MS;
  int*   knn = (int*)ws;    ws += sizeof(int)   * (size_t)BB * NP * KK;
  float* wc  = (float*)ws;  ws += sizeof(float) * (size_t)BB * NP * 3;
  float* lin = (float*)ws;  ws += sizeof(float) * (size_t)BB * NP;
  float* den = (float*)ws;  ws += sizeof(float) * (size_t)BB * NP;
  float* mm  = (float*)ws;

  fps_kernel   <<<BB,        1024, 0, stream>>>(points, idx, pts);
  knn_kernel   <<<BB*KT,      256, 0, stream>>>(pts, knn);
  geom_kernel  <<<(BB*NP+255)/256, 256, 0, stream>>>(pts, knn, wc, lin, den);
  mlp_wmma_kernel<<<BB*MT,     32, 0, stream>>>(wc, pts, rw1, rb1, rw2, rb2,
                                                sw1, sb1, sw2, sb2, cov, Rm, scl);
  minmax_kernel<<<BB,         256, 0, stream>>>(lin, den, mm);
  opacity_kernel<<<(BB*NP+255)/256, 256, 0, stream>>>(lin, den, mm, opa);
}
